// superglue_72370198937924
// MI455X (gfx1250) — matrix operand order repaired and checked by reading, not compiled, not run on hardware
//
#include <hip/hip_runtime.h>
#include <hip/hip_bf16.h>
#include <math.h>

typedef __attribute__((ext_vector_type(16))) _Float16 v16h;
typedef __attribute__((ext_vector_type(8)))  float    v8f;

#define NPTS   192
#define CH     128
#define NBATCH 4
#define TOK    384            // 2*NPTS
#define SCALE_F 11.313708498984761f
#define REG_F   0.01f
#define SINK_ITERS 100

// ---------------------------------------------------------------------------
// Generic WMMA GEMM:  C[b,m,n] = (Σ_k A[b,m,k] * Bt[b,n,k]) + bias_scale*bias[n]
//                              (+ residual[b,m,n]) (+ optional relu)
// K is a compile-time constant -> fully unrolled K-loop -> K/32 back-to-back
// v_wmma_f32_16x16x32_f16 per wave with float4 (b128) fragment loads.
// Block = 32x4: four independent waves, each owns one 16x16 tile along M.
// All fragment addresses are 16B aligned (rows are 512B/1KB, offsets 8/16 fl).
// ---------------------------------------------------------------------------
template <int K>
__global__ __launch_bounds__(128)
void k_gemm_wmma(const float* __restrict__ A,  int lda, size_t batchA,
                 const float* __restrict__ Bt, int ldb, size_t batchB,
                 const float* __restrict__ bias, float bias_scale,
                 const float* __restrict__ residual,
                 float* __restrict__ C, int ldc, size_t batchC,
                 int do_relu)
{
    const int tm   = blockIdx.x * 4 + threadIdx.y;   // 16-row tile index
    const int tn   = blockIdx.y;
    const int b    = blockIdx.z;
    const int lane = threadIdx.x;
    const int half = (lane >> 4) & 1;   // lane group 0-15 vs 16-31
    const int ar   = lane & 15;         // A row / B col within tile

    const float* Ab = A  + batchA * b + (size_t)(tm * 16 + ar) * lda;
    const float* Bb = Bt + batchB * b + (size_t)(tn * 16 + ar) * ldb;

    v8f acc = {};
    #pragma unroll
    for (int k0 = 0; k0 < K; k0 += 32) {
        // A fragment (16x32 f16): lanes 0-15 hold K k0+0..7 / k0+16..23,
        // lanes 16-31 hold K k0+8..15 / k0+24..31 (row = lane & 15)
        const float4* ap = (const float4*)(Ab + k0 + (half ? 8 : 0));
        const float4 a0 = ap[0], a1 = ap[1];   // elems 0..7
        const float4 a2 = ap[4], a3 = ap[5];   // elems +16..+23
        // B fragment (32x16 f16): col = lane & 15; lanes 0-15 hold K k0+0..7 / k0+16..23,
        // lanes 16-31 hold K k0+8..15 / k0+24..31.  Bt is row-major (N x K).
        const float4* bp = (const float4*)(Bb + k0 + (half ? 8 : 0));
        const float4 b0 = bp[0], b1 = bp[1], b2 = bp[4], b3 = bp[5];

        v16h av, bv;
        av[0]  = (_Float16)a0.x; av[1]  = (_Float16)a0.y;
        av[2]  = (_Float16)a0.z; av[3]  = (_Float16)a0.w;
        av[4]  = (_Float16)a1.x; av[5]  = (_Float16)a1.y;
        av[6]  = (_Float16)a1.z; av[7]  = (_Float16)a1.w;
        av[8]  = (_Float16)a2.x; av[9]  = (_Float16)a2.y;
        av[10] = (_Float16)a2.z; av[11] = (_Float16)a2.w;
        av[12] = (_Float16)a3.x; av[13] = (_Float16)a3.y;
        av[14] = (_Float16)a3.z; av[15] = (_Float16)a3.w;

        bv[0]  = (_Float16)b0.x; bv[1]  = (_Float16)b0.y;
        bv[2]  = (_Float16)b0.z; bv[3]  = (_Float16)b0.w;
        bv[4]  = (_Float16)b1.x; bv[5]  = (_Float16)b1.y;
        bv[6]  = (_Float16)b1.z; bv[7]  = (_Float16)b1.w;
        bv[8]  = (_Float16)b2.x; bv[9]  = (_Float16)b2.y;
        bv[10] = (_Float16)b2.z; bv[11] = (_Float16)b2.w;
        bv[12] = (_Float16)b3.x; bv[13] = (_Float16)b3.y;
        bv[14] = (_Float16)b3.z; bv[15] = (_Float16)b3.w;

        acc = __builtin_amdgcn_wmma_f32_16x16x32_f16(
                  false, av, false, bv, (short)0, acc, false, false);
    }

    const int col = tn * 16 + ar;
    const float bv0 = bias ? bias_scale * bias[col] : 0.0f;
    #pragma unroll
    for (int r = 0; r < 8; ++r) {
        const int row = tm * 16 + r + (half ? 8 : 0);
        float o = acc[r] + bv0;
        if (residual) o += residual[batchC * b + (size_t)row * ldc + col];
        if (do_relu)  o = fmaxf(o, 0.0f);
        C[batchC * b + (size_t)row * ldc + col] = o;
    }
}

// ---------------------------------------------------------------------------
// Positional encoder: x[b,t,:] = relu(fc2(relu(fc1(p)))) + d
// grid (TOK, NBATCH), block 128 (one thread per output channel)
// ---------------------------------------------------------------------------
__global__ __launch_bounds__(128)
void k_posenc(const float* __restrict__ p1, const float* __restrict__ d1,
              const float* __restrict__ p2, const float* __restrict__ d2,
              const float* __restrict__ W1, const float* __restrict__ b1,
              const float* __restrict__ W2, const float* __restrict__ b2,
              float* __restrict__ x)
{
    const int t = blockIdx.x;   // token 0..383
    const int b = blockIdx.y;
    const int c = threadIdx.x;  // channel 0..127
    __shared__ float hid[64];

    const int   tn = (t < NPTS) ? t : t - NPTS;
    const float* p = (t < NPTS) ? p1 + ((size_t)b * NPTS + tn) * 2
                                : p2 + ((size_t)b * NPTS + tn) * 2;
    const float* d = (t < NPTS) ? d1 + ((size_t)b * NPTS + tn) * CH
                                : d2 + ((size_t)b * NPTS + tn) * CH;
    if (c < 64) {
        float h = W1[c * 2] * p[0] + W1[c * 2 + 1] * p[1] + b1[c];
        hid[c] = fmaxf(h, 0.0f);
    }
    __syncthreads();
    float o = b2[c];
    #pragma unroll 8
    for (int h = 0; h < 64; ++h) o += W2[c * 64 + h] * hid[h];
    o = fmaxf(o, 0.0f) + d[c];
    x[((size_t)b * TOK + t) * CH + c] = o;
}

// ---------------------------------------------------------------------------
// Attentional aggregation (linearity-folded):
//   s[b,tgt,c] = Σ_src softmax_c(q[tgt,c]*k[src,c]/SCALE) * v[src,c]
// Dense edges: intra = same image, src != tgt; cross = other image.
// One wave per (tgt, batch); 4 channels per lane; wave32 shuffle reductions.
// global_prefetch pulls the next src rows of k/v while reducing the current.
// ---------------------------------------------------------------------------
__global__ __launch_bounds__(32)
void k_attconv(const float* __restrict__ q, const float* __restrict__ k,
               const float* __restrict__ v, float* __restrict__ s, int cross)
{
    const int tgt  = blockIdx.x;           // 0..383
    const int b    = blockIdx.y;
    const int lane = threadIdx.x;          // 0..31
    const size_t bs = (size_t)TOK * CH;
    const float* qb = q + b * bs;
    const float* kb = k + b * bs;
    const float* vb = v + b * bs;

    float qv[4], acc[4] = {0.f, 0.f, 0.f, 0.f};
    #pragma unroll
    for (int r = 0; r < 4; ++r)
        qv[r] = qb[(size_t)tgt * CH + lane + 32 * r] * (1.0f / SCALE_F);

    const int half     = tgt / NPTS;                       // which image
    const int src_base = (cross ? (1 - half) : half) * NPTS;

    for (int ss = 0; ss < NPTS; ++ss) {
        const int src = src_base + ss;
        // speculative prefetch of upcoming src rows (global_prefetch_b8)
        __builtin_prefetch(kb + (size_t)(src + 4) * CH + lane * 4, 0, 1);
        __builtin_prefetch(vb + (size_t)(src + 4) * CH + lane * 4, 0, 1);
        if (!cross && src == tgt) continue;
        float e[4];
        float m = -3.0e38f;
        #pragma unroll
        for (int r = 0; r < 4; ++r) {
            e[r] = qv[r] * kb[(size_t)src * CH + lane + 32 * r];
            m = fmaxf(m, e[r]);
        }
        #pragma unroll
        for (int off = 16; off >= 1; off >>= 1)
            m = fmaxf(m, __shfl_xor(m, off, 32));
        float sum = 0.f;
        #pragma unroll
        for (int r = 0; r < 4; ++r) { e[r] = __expf(e[r] - m); sum += e[r]; }
        #pragma unroll
        for (int off = 16; off >= 1; off >>= 1)
            sum += __shfl_xor(sum, off, 32);
        const float inv = 1.0f / sum;
        #pragma unroll
        for (int r = 0; r < 4; ++r)
            acc[r] += e[r] * inv * vb[(size_t)src * CH + lane + 32 * r];
    }
    #pragma unroll
    for (int r = 0; r < 4; ++r)
        s[b * bs + (size_t)tgt * CH + lane + 32 * r] = acc[r];
}

// ---------------------------------------------------------------------------
// cat = [x, a] along channels (B,TOK,256)
// ---------------------------------------------------------------------------
__global__ void k_concat(const float* __restrict__ x, const float* __restrict__ a,
                         float* __restrict__ cat)
{
    const size_t idx = (size_t)blockIdx.x * blockDim.x + threadIdx.x;
    const size_t tot = (size_t)NBATCH * TOK * 2 * CH;
    if (idx >= tot) return;
    const size_t tok = idx / (2 * CH);
    const int    ch  = (int)(idx % (2 * CH));
    cat[idx] = (ch < CH) ? x[tok * CH + ch] : a[tok * CH + (ch - CH)];
}

// ---------------------------------------------------------------------------
// Row-wise L2 normalization of x5 (B,TOK,CH). grid (TOK,B), block 128.
// ---------------------------------------------------------------------------
__global__ __launch_bounds__(128)
void k_normalize(float* __restrict__ x5)
{
    const int t = blockIdx.x, b = blockIdx.y, c = threadIdx.x;
    __shared__ float red[128];
    const size_t base = ((size_t)b * TOK + t) * CH;
    const float val = x5[base + c];
    red[c] = val * val;
    __syncthreads();
    #pragma unroll
    for (int st = 64; st > 0; st >>= 1) {
        if (c < st) red[c] += red[c + st];
        __syncthreads();
    }
    x5[base + c] = val * rsqrtf(red[0]);
}

// ---------------------------------------------------------------------------
// Mr[b,i,j] = (cw - 1)/reg  with dustbin row/col.  (Mr = -M/reg)
// ---------------------------------------------------------------------------
__global__ void k_mrinit(const float* __restrict__ costs,
                         const float* __restrict__ dustbin,
                         float* __restrict__ Mr)
{
    const int LD  = NPTS + 1;                  // 193
    const size_t idx = (size_t)blockIdx.x * blockDim.x + threadIdx.x;
    const size_t tot = (size_t)NBATCH * LD * LD;
    if (idx >= tot) return;
    const int b = (int)(idx / (LD * LD));
    const int i = (int)((idx / LD) % LD);
    const int j = (int)(idx % LD);
    const float w = dustbin[0];
    const float cw = (i < NPTS && j < NPTS)
                   ? costs[((size_t)b * NPTS + i) * NPTS + j] : w;
    Mr[idx] = (cw - 1.0f) * (1.0f / REG_F);
}

// ---------------------------------------------------------------------------
// Sinkhorn (log domain), one workgroup per batch, f/g in LDS.
// out[b,i,j] = exp(Mr + f[i]/reg + g[j]/reg)
// ---------------------------------------------------------------------------
__global__ __launch_bounds__(256)
void k_sinkhorn(const float* __restrict__ Mr, float* __restrict__ out)
{
    const int LD = NPTS + 1;                   // 193
    const int b   = blockIdx.x;
    const int tid = threadIdx.x;
    __shared__ float fsh[NPTS + 1];
    __shared__ float gsh[NPTS + 1];
    const float* MrB = Mr + (size_t)b * LD * LD;
    const float invreg = 1.0f / REG_F;
    const float logn   = logf((float)NPTS);

    if (tid < LD) { fsh[tid] = 0.f; gsh[tid] = 0.f; }
    __syncthreads();

    for (int it = 0; it < SINK_ITERS; ++it) {
        if (tid < LD) {                         // f update: LSE over j
            const float* row = MrB + (size_t)tid * LD;
            float m = -3.0e38f, sacc = 0.f;
            for (int j = 0; j < LD; ++j) {
                const float val = row[j] + gsh[j] * invreg;
                if (val > m) { sacc = sacc * __expf(m - val) + 1.0f; m = val; }
                else         { sacc += __expf(val - m); }
            }
            const float loga = (tid == NPTS) ? logn : 0.0f;
            fsh[tid] = REG_F * (loga - (m + logf(sacc)));
        }
        __syncthreads();
        if (tid < LD) {                         // g update: LSE over i
            float m = -3.0e38f, sacc = 0.f;
            for (int i = 0; i < LD; ++i) {
                const float val = MrB[(size_t)i * LD + tid] + fsh[i] * invreg;
                if (val > m) { sacc = sacc * __expf(m - val) + 1.0f; m = val; }
                else         { sacc += __expf(val - m); }
            }
            const float logb = (tid == NPTS) ? logn : 0.0f;
            gsh[tid] = REG_F * (logb - (m + logf(sacc)));
        }
        __syncthreads();
    }

    for (int idx = tid; idx < LD * LD; idx += 256) {
        const int i = idx / LD, j = idx % LD;
        out[(size_t)b * LD * LD + idx] =
            __expf(MrB[idx] + fsh[i] * invreg + gsh[j] * invreg);
    }
}

// ---------------------------------------------------------------------------
extern "C" void kernel_launch(void* const* d_in, const int* in_sizes, int n_in,
                              void* d_out, int out_size, void* d_ws, size_t ws_size,
                              hipStream_t stream)
{
    (void)in_sizes; (void)n_in; (void)out_size; (void)ws_size;

    // Input order: setup_inputs() dict order; params in insertion order,
    // each (W, b) flattened as W then b.
    const float* p1 = (const float*)d_in[0];
    const float* d1 = (const float*)d_in[1];
    const float* p2 = (const float*)d_in[2];
    const float* d2 = (const float*)d_in[3];
    // d_in[4..6]: matches / edges (dense structure known, unused)
    const float* fc1W = (const float*)d_in[7];
    const float* fc1b = (const float*)d_in[8];
    const float* fc2W = (const float*)d_in[9];
    const float* fc2b = (const float*)d_in[10];
    const float* fc3W = (const float*)d_in[51];
    const float* fc3b = (const float*)d_in[52];
    const float* dust = (const float*)d_in[53];

    // Workspace carve-out (floats)
    float* ws = (float*)d_ws;
    const size_t SX = (size_t)NBATCH * TOK * CH;     // 196608
    float* x     = ws;              // (B,384,128)
    float* q     = x     + SX;
    float* kbuf  = q     + SX;
    float* v     = kbuf  + SX;
    float* s     = v     + SX;
    float* a     = s     + SX;
    float* cat   = a     + SX;                       // (B,384,256)
    float* x5    = cat   + 2 * SX;
    float* costs = x5    + SX;                       // (B,192,192)
    float* Mr    = costs + (size_t)NBATCH * NPTS * NPTS; // (B,193,193)
    float* out   = (float*)d_out;

    const size_t bsX   = (size_t)TOK * CH;
    const size_t bsCat = (size_t)TOK * 2 * CH;

    const dim3 gblk(128);                 // 4 waves per workgroup
    const dim3 g128(TOK / 64, CH / 16, NBATCH);

    // 1) positional encoding + descriptor add
    k_posenc<<<dim3(TOK, NBATCH), 128, 0, stream>>>(
        p1, d1, p2, d2, fc1W, fc1b, fc2W, fc2b, x);

    // 2) attentional blocks
    for (int m = 1; m <= 4; ++m) {
        const int base = 11 + (m - 1) * 10;
        const float* Wq = (const float*)d_in[base + 0];
        const float* bq = (const float*)d_in[base + 1];
        const float* Wk = (const float*)d_in[base + 2];
        const float* bk = (const float*)d_in[base + 3];
        const float* Wv = (const float*)d_in[base + 4];
        const float* bv = (const float*)d_in[base + 5];
        const float* Wo = (const float*)d_in[base + 6];
        const float* bo = (const float*)d_in[base + 7];
        const float* Wm = (const float*)d_in[base + 8];
        const float* bm = (const float*)d_in[base + 9];
        const int cross = (m % 2 == 0);
        const float deg = cross ? (float)NPTS : (float)(NPTS - 1);

        k_gemm_wmma<CH><<<g128, gblk, 0, stream>>>(x, CH, bsX, Wq, CH, 0,
            bq, 1.0f, nullptr, q,    CH, bsX, 0);
        k_gemm_wmma<CH><<<g128, gblk, 0, stream>>>(x, CH, bsX, Wk, CH, 0,
            bk, 1.0f, nullptr, kbuf, CH, bsX, 0);
        k_gemm_wmma<CH><<<g128, gblk, 0, stream>>>(x, CH, bsX, Wv, CH, 0,
            bv, 1.0f, nullptr, v,    CH, bsX, 0);

        k_attconv<<<dim3(TOK, NBATCH), 32, 0, stream>>>(q, kbuf, v, s, cross);

        // o-projection folded over the segment sum (bias scaled by degree)
        k_gemm_wmma<CH><<<g128, gblk, 0, stream>>>(s, CH, bsX, Wo, CH, 0,
            bo, deg, nullptr, a, CH, bsX, 0);

        // mlp over concat([x, a]) with residual into x
        const size_t catTot = (size_t)NBATCH * TOK * 2 * CH;
        k_concat<<<(unsigned)((catTot + 255) / 256), 256, 0, stream>>>(x, a, cat);
        k_gemm_wmma<2 * CH><<<g128, gblk, 0, stream>>>(cat, 2 * CH, bsCat,
            Wm, 2 * CH, 0, bm, 1.0f, x, x, CH, bsX, 0);
    }

    // 3) fc3 + relu, then L2 normalize rows
    k_gemm_wmma<CH><<<g128, gblk, 0, stream>>>(x, CH, bsX, fc3W, CH, 0,
        fc3b, 1.0f, nullptr, x5, CH, bsX, 1);
    k_normalize<<<dim3(TOK, NBATCH), 128, 0, stream>>>(x5);

    // 4) cost matrix: v1 @ v2^T (Bt operand = v2 rows, row-major N x K)
    const dim3 gcost(NPTS / 64, NPTS / 16, NBATCH);
    k_gemm_wmma<CH><<<gcost, gblk, 0, stream>>>(
        x5, CH, bsX,                              // A  = v1 (first 192 rows)
        x5 + (size_t)NPTS * CH, CH, bsX,          // Bt = v2 (rows 192..383)
        nullptr, 0.0f, nullptr,
        costs, NPTS, (size_t)NPTS * NPTS, 0);

    // 5) dustbin-augmented Mr, then Sinkhorn -> transport plan
    const size_t mrTot = (size_t)NBATCH * (NPTS + 1) * (NPTS + 1);
    k_mrinit<<<(unsigned)((mrTot + 255) / 256), 256, 0, stream>>>(costs, dust, Mr);
    k_sinkhorn<<<NBATCH, 256, 0, stream>>>(Mr, out);
}
